// RNNLayer_64312840290574
// MI455X (gfx1250) — compile-verified, matrix operand order checked by reading, not run on hardware
//
#include <hip/hip_runtime.h>

// ---------------------------------------------------------------------------
// RNN layer for MI455X (gfx1250):
//   u  = x @ b            (bf16 WMMA GEMM, f32 accum)
//   hs = scan(h = h*a + u) (f32 sequential scan, 8192 parallel chains)
//   y  = hs @ c           (bf16 WMMA GEMM, f32 accum)
// GEMM: 64x64 per wave (4x4 v_wmma_f32_16x16x32_bf16 tiles), 128x128 per block.
// ---------------------------------------------------------------------------

#define BS   8
#define T    4096
#define D_H  1024
#define M_TOTAL (BS * T)          // 32768 rows

typedef __attribute__((ext_vector_type(8)))  float   v8f;
typedef __attribute__((ext_vector_type(8)))  __bf16  bf16x8;
typedef __attribute__((ext_vector_type(16))) __bf16  bf16x16;

// RTNE float -> bf16 without relying on fptrunc codegen
__device__ __forceinline__ __bf16 f32_to_bf16(float f) {
    union { float f; unsigned u; } in;
    in.f = f;
    unsigned r = in.u + 0x7fffu + ((in.u >> 16) & 1u);
    unsigned short h = (unsigned short)(r >> 16);
    return __builtin_bit_cast(__bf16, h);
}

// ---------------------------------------------------------------------------
// Kernel 1: elementwise f32 -> bf16 (8 elems / thread, b128 loads, b128 store)
// ---------------------------------------------------------------------------
__global__ __launch_bounds__(256)
void cvt_f32_bf16_kernel(const float* __restrict__ in, __bf16* __restrict__ out, long n) {
    long i = ((long)blockIdx.x * blockDim.x + threadIdx.x) * 8;
    if (i >= n) return;
    float4 f0 = *(const float4*)(in + i);
    float4 f1 = *(const float4*)(in + i + 4);
    bf16x8 o;
    o[0] = f32_to_bf16(f0.x); o[1] = f32_to_bf16(f0.y);
    o[2] = f32_to_bf16(f0.z); o[3] = f32_to_bf16(f0.w);
    o[4] = f32_to_bf16(f1.x); o[5] = f32_to_bf16(f1.y);
    o[6] = f32_to_bf16(f1.z); o[7] = f32_to_bf16(f1.w);
    *(bf16x8*)(out + i) = o;
}

// ---------------------------------------------------------------------------
// Kernel 2: 32x32 LDS-tiled transpose + convert: out[c*R + r] = bf16(in[r*C + c])
// blockDim (32,8); grid (C/32, R/32)
// ---------------------------------------------------------------------------
__global__ __launch_bounds__(256)
void transpose_cvt_kernel(const float* __restrict__ in, __bf16* __restrict__ out,
                          int R, int C) {
    __shared__ float tile[32][33];
    int c0 = blockIdx.x * 32;
    int r0 = blockIdx.y * 32;
    int tx = threadIdx.x;
    int ty = threadIdx.y;
#pragma unroll
    for (int i = 0; i < 32; i += 8)
        tile[ty + i][tx] = in[(size_t)(r0 + ty + i) * C + (c0 + tx)];
    __syncthreads();
#pragma unroll
    for (int i = 0; i < 32; i += 8)
        out[(size_t)(c0 + ty + i) * R + (r0 + tx)] = f32_to_bf16(tile[tx][ty + i]);
}

// ---------------------------------------------------------------------------
// WMMA fragment load: 16-bit A-matrix 16x32 layout (ISA 7.12.2).
// Lane L (L<16):  row M=L,  K = {k0..k0+7, k0+16..k0+23}
// Lane L+16:      row M=L,  K = {k0+8..k0+15, k0+24..k0+31}
// -> two contiguous 16-byte loads per lane. B fragments are identical on Bt
//    (column-major B), since the B layout is A's layout of B^T.
// ---------------------------------------------------------------------------
__device__ __forceinline__ bf16x16 load_frag(const __bf16* p) {
    bf16x8 lo = *(const bf16x8*)(p);
    bf16x8 hi = *(const bf16x8*)(p + 16);
    return __builtin_shufflevector(lo, hi, 0, 1, 2, 3, 4, 5, 6, 7,
                                           8, 9, 10, 11, 12, 13, 14, 15);
}

// ---------------------------------------------------------------------------
// Kernel 3: GEMM  C[M,N] = A[M,K] * Bt[N,K]^T   (bf16 in, f32 out)
// 128 threads = 4 waves in a 2x2 grid; each wave owns a 64x64 output tile
// (4x4 WMMA 16x16 tiles, 16 v8f accumulators). Block tile 128x128.
// Per K=32 step: 16 b128 loads feed 16 WMMAs (1 WMMA per load).
// grid = (N/128, M/128).
// ---------------------------------------------------------------------------
__global__ __launch_bounds__(128)
void gemm_bf16_wmma_kernel(const __bf16* __restrict__ A,
                           const __bf16* __restrict__ Bt,
                           float* __restrict__ C,
                           int M, int N, int K) {
    const int lane = threadIdx.x & 31;
    const int wave = threadIdx.x >> 5;
    const int wm = wave >> 1;          // 0..1
    const int wn = wave & 1;           // 0..1
    const int bm = blockIdx.y * 128 + wm * 64;
    const int bn = blockIdx.x * 128 + wn * 64;

    const int r    = lane & 15;
    const int koff = (lane >> 4) << 3;   // 0 or 8

    const __bf16* ap[4];
    const __bf16* bp[4];
#pragma unroll
    for (int i = 0; i < 4; ++i) {
        ap[i] = A  + (size_t)(bm + 16 * i + r) * K + koff;
        bp[i] = Bt + (size_t)(bn + 16 * i + r) * K + koff;
    }

    v8f acc[4][4];
#pragma unroll
    for (int i = 0; i < 4; ++i)
#pragma unroll
        for (int j = 0; j < 4; ++j)
            acc[i][j] = (v8f){};

    for (int k = 0; k < K; k += 32) {
        bf16x16 fa[4], fb[4];
#pragma unroll
        for (int i = 0; i < 4; ++i) fa[i] = load_frag(ap[i] + k);
#pragma unroll
        for (int j = 0; j < 4; ++j) fb[j] = load_frag(bp[j] + k);
#pragma unroll
        for (int i = 0; i < 4; ++i)
#pragma unroll
            for (int j = 0; j < 4; ++j)
                acc[i][j] = __builtin_amdgcn_wmma_f32_16x16x32_bf16(
                    false, fa[i], false, fb[j], (short)0, acc[i][j], false, false);
    }

    // C/D layout: VGPR v, lane: n = lane&15, m = v + 8*(lane>>4)
    const int n   = lane & 15;
    const int mhi = (lane >> 4) << 3;    // 0 or 8
#pragma unroll
    for (int i = 0; i < 4; ++i) {
        float* crow = C + (size_t)(bm + 16 * i + mhi) * N + bn + n;
#pragma unroll
        for (int j = 0; j < 4; ++j)
#pragma unroll
            for (int v = 0; v < 8; ++v)
                crow[(size_t)v * N + 16 * j] = acc[i][j][v];
    }
}

// ---------------------------------------------------------------------------
// Kernel 4: diagonal linear recurrence h = h*a + u over T, per (batch, channel).
// 8192 independent chains; reads f32 u, writes bf16 hs (feed to GEMM2).
// Loads unrolled x4 so 4 independent global loads are in flight per FMA chain.
// ---------------------------------------------------------------------------
__global__ __launch_bounds__(256)
void scan_kernel(const float* __restrict__ u, const float* __restrict__ a,
                 __bf16* __restrict__ hs) {
    int idx = blockIdx.x * blockDim.x + threadIdx.x;
    if (idx >= BS * D_H) return;
    int b = idx >> 10;           // / D_H
    int h = idx & (D_H - 1);

    const float av = a[h];
    float s = 0.0f;
    const float* up = u  + (size_t)b * T * D_H + h;
    __bf16*      op = hs + (size_t)b * T * D_H + h;

    for (int t = 0; t < T; t += 4) {
        float u0 = up[(size_t)(t + 0) * D_H];
        float u1 = up[(size_t)(t + 1) * D_H];
        float u2 = up[(size_t)(t + 2) * D_H];
        float u3 = up[(size_t)(t + 3) * D_H];
        s = fmaf(s, av, u0); op[(size_t)(t + 0) * D_H] = f32_to_bf16(s);
        s = fmaf(s, av, u1); op[(size_t)(t + 1) * D_H] = f32_to_bf16(s);
        s = fmaf(s, av, u2); op[(size_t)(t + 2) * D_H] = f32_to_bf16(s);
        s = fmaf(s, av, u3); op[(size_t)(t + 3) * D_H] = f32_to_bf16(s);
    }
}

// ---------------------------------------------------------------------------
// Launch: cvt(x) ; transpose(b) ; transpose(c) ; GEMM1 ; scan ; GEMM2
// Workspace layout (196 MB):
//   [0,      64MB)  x_bf16  -> reused as hs_bf16 after GEMM1
//   [64MB,   66MB)  Bt bf16 (1024x1024, col-major B)
//   [66MB,   68MB)  Ct bf16
//   [68MB,  196MB)  u  f32  (32768x1024)
// ---------------------------------------------------------------------------
extern "C" void kernel_launch(void* const* d_in, const int* in_sizes, int n_in,
                              void* d_out, int out_size, void* d_ws, size_t ws_size,
                              hipStream_t stream) {
    const float* x = (const float*)d_in[0];   // [8,4096,1024]
    const float* a = (const float*)d_in[1];   // [1024]
    const float* b = (const float*)d_in[2];   // [1024,1024]
    const float* c = (const float*)d_in[3];   // [1024,1024]
    float*       y = (float*)d_out;           // [8,4096,1024]

    char* ws = (char*)d_ws;
    const size_t MB = 1024ull * 1024ull;
    __bf16* xbf = (__bf16*)(ws);                 // 64 MB
    __bf16* Bt  = (__bf16*)(ws + 64 * MB);       //  2 MB
    __bf16* Ct  = (__bf16*)(ws + 66 * MB);       //  2 MB
    float*  u   = (float*) (ws + 68 * MB);       // 128 MB
    __bf16* hs  = xbf;                           // alias: x_bf16 dead after GEMM1

    const long n_x = (long)M_TOTAL * D_H;        // 33554432

    // 1) x -> bf16
    cvt_f32_bf16_kernel<<<(unsigned)(n_x / 8 / 256), 256, 0, stream>>>(x, xbf, n_x);

    // 2,3) b, c -> transposed bf16 (column-major)
    dim3 tb(32, 8), tg(D_H / 32, D_H / 32);
    transpose_cvt_kernel<<<tg, tb, 0, stream>>>(b, Bt, D_H, D_H);
    transpose_cvt_kernel<<<tg, tb, 0, stream>>>(c, Ct, D_H, D_H);

    // 4) u = x @ b
    dim3 gblk(128), ggrd(D_H / 128, M_TOTAL / 128);   // (8, 256)
    gemm_bf16_wmma_kernel<<<ggrd, gblk, 0, stream>>>(xbf, Bt, u, M_TOTAL, D_H, D_H);

    // 5) diagonal scan -> hs (bf16)
    scan_kernel<<<(BS * D_H) / 256, 256, 0, stream>>>(u, a, hs);

    // 6) y = hs @ c
    gemm_bf16_wmma_kernel<<<ggrd, gblk, 0, stream>>>(hs, Ct, y, M_TOTAL, D_H, D_H);
}
